// LSTMRegressor_9045201125842
// MI455X (gfx1250) — compile-verified, matrix operand order checked by reading, not run on hardware
//
#include <hip/hip_runtime.h>

// LSTM fused 2-layer + FC head for MI455X (gfx1250, wave32, WMMA).
// B=4096, T=256, I=8, H=64, gates=256.
//
// Cross-layer software pipeline: phase k runs layer0(t=k) on waves 0-3 and
// layer1(t=k-1) on waves 4-7 concurrently. Each wave owns the 4 gate tiles
// (i,f,g,o) of one 16-wide hh slice, so the LSTM cell update happens directly
// on the WMMA accumulators in registers -> no gate scratch in LDS and only
// ONE barrier per phase. Hidden states are double-buffered bf16 in LDS.
// Recurrent weights (Whh0/Whh1) live as WMMA B-fragments in VGPRs.
// Accumulators are zero-seeded (inline-0 C operand); biases fold into the
// tanh-based gate math at zero extra cost. x is staged once to LDS as bf16.

#define TB 16       // batch rows per workgroup
#define HD 64       // hidden size
#define TSTEPS 256
#define IDIM 8
#define XROW (TSTEPS * IDIM)   // 2048 elements per batch row

typedef __attribute__((ext_vector_type(16))) __bf16        v16bf;
typedef __attribute__((ext_vector_type(8)))  float         v8f;
typedef __attribute__((ext_vector_type(4)))  unsigned int  v4u;

union FragU {
    v16bf          bf;
    v4u            u[2];
    unsigned short s[16];
};

__device__ __forceinline__ unsigned short f2bf(float x) {
    unsigned int u = __float_as_uint(x);
    u += 0x7FFFu + ((u >> 16) & 1u);          // round-to-nearest-even
    return (unsigned short)(u >> 16);
}
__device__ __forceinline__ float bf2f(unsigned short s) {
    return __uint_as_float(((unsigned int)s) << 16);
}

#if __has_builtin(__builtin_amdgcn_tanhf)
__device__ __forceinline__ float tanh_hw(float x) { return __builtin_amdgcn_tanhf(x); }
#elif __has_builtin(__builtin_amdgcn_tanh_f32)
__device__ __forceinline__ float tanh_hw(float x) { return __builtin_amdgcn_tanh_f32(x); }
#else
__device__ __forceinline__ float tanh_hw(float x) {
    float e = __expf(-2.0f * x);
    return (1.0f - e) / (1.0f + e);
}
#endif
// sigmoid(x + b) with hb = 0.5*b precomputed: 0.5*tanh(0.5x + hb) + 0.5
__device__ __forceinline__ float sigm_b(float x, float hb) {
    return __builtin_fmaf(0.5f, tanh_hw(__builtin_fmaf(0.5f, x, hb)), 0.5f);
}

__device__ __forceinline__ v8f wmma_bf16(v16bf a, v16bf b, v8f c) {
    return __builtin_amdgcn_wmma_f32_16x16x32_bf16(false, a, false, b,
                                                   (short)0, c, false, false);
}

// A fragment (16x32 bf16 slice) from row-major [16][64] bf16 LDS buffer.
__device__ __forceinline__ v16bf load_A64(const unsigned short* buf, int lane, int ko) {
    int row = lane & 15;
    int kb  = ko + ((lane >> 4) << 3);
    FragU f;
    f.u[0] = *(const v4u*)(buf + row * HD + kb);
    f.u[1] = *(const v4u*)(buf + row * HD + kb + 16);
    return f.bf;
}
// x A-fragment: only the K=0..7 positions of lanes 0-15 matter; all other
// positions multiply zero rows of the (zero-padded) Wih0 B-fragments, so any
// value is fine there -> a single ds_load_b128, no zero padding needed.
__device__ __forceinline__ v16bf load_Ax(const unsigned short* sx, int lane, int t) {
    int row = lane & 15;
    FragU f;
    f.u[0] = *(const v4u*)(sx + row * XROW + t * IDIM);
    f.u[1] = f.u[0];
    return f.bf;
}
// B fragment (32x16 bf16) from LDS weight matrix stored [256][64] gate-major.
__device__ __forceinline__ v16bf load_B64(const unsigned short* w, int lane, int nbase, int ko) {
    int n  = nbase + (lane & 15);
    int kb = ko + ((lane >> 4) << 3);
    FragU f;
    f.u[0] = *(const v4u*)(w + n * HD + kb);
    f.u[1] = *(const v4u*)(w + n * HD + kb + 16);
    return f.bf;
}
// B fragment built once from a global fp32 weight matrix [256][64].
__device__ __forceinline__ v16bf ldB_global(const float* __restrict__ w, int n, int kbase) {
    FragU f;
#pragma unroll
    for (int e = 0; e < 8; ++e) {
        f.s[e]     = f2bf(w[n * HD + kbase + e]);
        f.s[e + 8] = f2bf(w[n * HD + kbase + 16 + e]);
    }
    return f.bf;
}

__global__ __launch_bounds__(256)
void lstm_fused_kernel(const float* __restrict__ x,
                       const float* __restrict__ Wih0, const float* __restrict__ Whh0,
                       const float* __restrict__ bih0, const float* __restrict__ bhh0,
                       const float* __restrict__ Wih1, const float* __restrict__ Whh1,
                       const float* __restrict__ bih1, const float* __restrict__ bhh1,
                       const float* __restrict__ Wfc,  const float* __restrict__ bfc,
                       float* __restrict__ out)
{
    __shared__ __align__(16) unsigned short sX[TB * XROW];      // 64 KB x tile (bf16)
    __shared__ __align__(16) unsigned short sWih1[256 * HD];    // 32 KB
    __shared__ __align__(16) unsigned short sH1[2][TB * HD];    //  4 KB (ping-pong)
    __shared__ __align__(16) unsigned short sH2[2][TB * HD];    //  4 KB (ping-pong)

    const int tid  = threadIdx.x;
    const int lane = tid & 31;
    const int wave = tid >> 5;
    const int b0   = blockIdx.x * TB;

    const int grp  = wave >> 2;          // 0: layer-0 group, 1: layer-1 group
    const int w4   = wave & 3;           // wave index inside group -> hh slice
    const int ncol = lane & 15;
    const int kb   = (lane >> 4) << 3;   // per-lane K sub-offset inside a frag
    const int mrow = kb;                 // C/D: M = v + 8*(lane>=16)
    const int hh   = w4 * 16 + ncol;     // this lane's hidden column

    // ---- stage x tile (contiguous 128 KB global chunk) to LDS as bf16 ----
    {
        const float* xblk = x + (size_t)b0 * XROW;
        for (int e = tid * 4; e < TB * XROW; e += 256 * 4) {
            float4 v4 = *(const float4*)(xblk + e);
            unsigned int lo = (unsigned)f2bf(v4.x) | ((unsigned)f2bf(v4.y) << 16);
            unsigned int hi = (unsigned)f2bf(v4.z) | ((unsigned)f2bf(v4.w) << 16);
            *(unsigned int*)(sX + e)     = lo;
            *(unsigned int*)(sX + e + 2) = hi;
        }
    }
    // ---- stage Wih1 to LDS (bf16); zero both state buffers ----
    for (int i = tid; i < 256 * HD; i += 256) sWih1[i] = f2bf(Wih1[i]);
    for (int i = tid; i < TB * HD; i += 256) {
        sH1[0][i] = 0; sH1[1][i] = 0; sH2[0][i] = 0; sH2[1][i] = 0;
    }

    // ---- per-wave register weight fragments ----
    const float* Wr = grp ? Whh1 : Whh0;
    const float* bi = grp ? bih1 : bih0;
    const float* bh = grp ? bhh1 : bhh0;
    v16bf fWr[4][2];                     // recurrent weights [gate][ko]
    v16bf fWx[4];                        // Wih0 frags, K=8 zero-padded (group 0)
    float bs[4];
#pragma unroll
    for (int g = 0; g < 4; ++g) {
        const int n = (w4 + 4 * g) * 16 + ncol;
        bs[g] = bi[n] + bh[n];
#pragma unroll
        for (int koi = 0; koi < 2; ++koi)
            fWr[g][koi] = ldB_global(Wr, n, koi * 32 + kb);
        FragU fx;
#pragma unroll
        for (int e = 0; e < 16; ++e) fx.s[e] = 0;
        if (grp == 0 && lane < 16) {
#pragma unroll
            for (int e = 0; e < 8; ++e) fx.s[e] = f2bf(Wih0[n * IDIM + e]);
        }
        fWx[g] = fx.bf;
    }
    // bias folded into gate math: sigmoid gates use 0.5*b, tanh gate uses b
    const float hb0 = 0.5f * bs[0];
    const float hb1 = 0.5f * bs[1];
    const float b2  = bs[2];
    const float hb3 = 0.5f * bs[3];

    float cst[8];                        // cell state, fixed cells per lane
#pragma unroll
    for (int v = 0; v < 8; ++v) cst[v] = 0.0f;

    __syncthreads();

    // ================= phase loop: k in [0, T] =================
    // phase k: group0 computes h1(t=k); group1 computes h2(t=k-1).
    for (int k = 0; k <= TSTEPS; ++k) {
        if (grp == 0) {
            if (k < TSTEPS) {
                const unsigned short* hprev = sH1[(k + 1) & 1];  // h1(k-1)
                unsigned short*       hout  = sH1[k & 1];        // h1(k)

                v8f acc[4];
#pragma unroll
                for (int g = 0; g < 4; ++g)
                    acc[g] = (v8f){0.f, 0.f, 0.f, 0.f, 0.f, 0.f, 0.f, 0.f};

                v16bf ax = load_Ax(sX, lane, k);
#pragma unroll
                for (int g = 0; g < 4; ++g)
                    acc[g] = wmma_bf16(ax, fWx[g], acc[g]);
#pragma unroll
                for (int koi = 0; koi < 2; ++koi) {
                    v16bf ah = load_A64(hprev, lane, koi * 32);
#pragma unroll
                    for (int g = 0; g < 4; ++g)
                        acc[g] = wmma_bf16(ah, fWr[g][koi], acc[g]);
                }
                // in-register cell update (acc0..3 = i,f,g,o for this lane's hh)
#pragma unroll
                for (int v = 0; v < 8; ++v) {
                    float ig = sigm_b(acc[0][v], hb0);
                    float fg = sigm_b(acc[1][v], hb1);
                    float gg = tanh_hw(acc[2][v] + b2);
                    float og = sigm_b(acc[3][v], hb3);
                    float c  = fg * cst[v] + ig * gg;
                    cst[v]   = c;
                    hout[(mrow + v) * HD + hh] = f2bf(og * tanh_hw(c));
                }
            }
        } else {
            if (k >= 1) {
                const unsigned short* h1prev = sH1[(k + 1) & 1]; // h1(k-1)
                const unsigned short* h2prev = sH2[k & 1];       // h2(k-2)
                unsigned short*       hout   = sH2[(k + 1) & 1]; // h2(k-1)

                v8f acc[4];
#pragma unroll
                for (int g = 0; g < 4; ++g)
                    acc[g] = (v8f){0.f, 0.f, 0.f, 0.f, 0.f, 0.f, 0.f, 0.f};
#pragma unroll
                for (int koi = 0; koi < 2; ++koi) {
                    v16bf a1 = load_A64(h1prev, lane, koi * 32);
#pragma unroll
                    for (int g = 0; g < 4; ++g)
                        acc[g] = wmma_bf16(a1,
                                 load_B64(sWih1, lane, (w4 + 4 * g) * 16, koi * 32),
                                 acc[g]);
                    v16bf a2 = load_A64(h2prev, lane, koi * 32);
#pragma unroll
                    for (int g = 0; g < 4; ++g)
                        acc[g] = wmma_bf16(a2, fWr[g][koi], acc[g]);
                }
#pragma unroll
                for (int v = 0; v < 8; ++v) {
                    float ig = sigm_b(acc[0][v], hb0);
                    float fg = sigm_b(acc[1][v], hb1);
                    float gg = tanh_hw(acc[2][v] + b2);
                    float og = sigm_b(acc[3][v], hb3);
                    float c  = fg * cst[v] + ig * gg;
                    cst[v]   = c;
                    hout[(mrow + v) * HD + hh] = f2bf(og * tanh_hw(c));
                }
            }
        }
        __syncthreads();   // single barrier per phase
    }

    // ---- FC head: h2(T-1) lives in sH2[(T-1)&1] = sH2[1] ----
    if (tid < TB) {
        float s = bfc[0];
#pragma unroll
        for (int j = 0; j < HD; ++j)
            s += bf2f(sH2[1][tid * HD + j]) * Wfc[j];
        out[b0 + tid] = s;
    }
}

extern "C" void kernel_launch(void* const* d_in, const int* in_sizes, int n_in,
                              void* d_out, int out_size, void* d_ws, size_t ws_size,
                              hipStream_t stream) {
    const float* x    = (const float*)d_in[0];
    const float* Wih0 = (const float*)d_in[1];
    const float* Whh0 = (const float*)d_in[2];
    const float* bih0 = (const float*)d_in[3];
    const float* bhh0 = (const float*)d_in[4];
    const float* Wih1 = (const float*)d_in[5];
    const float* Whh1 = (const float*)d_in[6];
    const float* bih1 = (const float*)d_in[7];
    const float* bhh1 = (const float*)d_in[8];
    const float* Wfc  = (const float*)d_in[9];
    const float* bfc  = (const float*)d_in[10];
    float* out = (float*)d_out;

    dim3 grid(4096 / TB);   // 256 workgroups, one 16-row batch tile each
    dim3 block(256);        // 8 wave32 waves: 4 for layer0, 4 for layer1
    lstm_fused_kernel<<<grid, block, 0, stream>>>(x, Wih0, Whh0, bih0, bhh0,
                                                  Wih1, Whh1, bih1, bhh1,
                                                  Wfc, bfc, out);
}
